// SceneGraphModel_10307921510470
// MI455X (gfx1250) — matrix operand order repaired and checked by reading, not compiled, not run on hardware
//
#include <hip/hip_runtime.h>

#define BB 32
#define NN 128
#define HH 64

typedef __attribute__((ext_vector_type(16))) _Float16 v16h;
typedef __attribute__((ext_vector_type(8)))  float    v8f;

static __device__ __forceinline__ _Float16 reluh(float x) {
    return (_Float16)(x > 0.f ? x : 0.f);
}

// ---------------------------------------------------------------------------
// Kernel 0a: combined = feat + coords@Wp + bp ;  Ai' = combined@W1[:H] + b1 ;
//            Aj = combined@W1[H:].   One block (64 threads) per (b,n) row.
// ---------------------------------------------------------------------------
__global__ __launch_bounds__(64) void sg_prelayer(
    const float* __restrict__ feat, const float* __restrict__ coords,
    const float* __restrict__ Wp,   const float* __restrict__ bp,
    const float* __restrict__ W1,   const float* __restrict__ b1,
    float* __restrict__ Ai, float* __restrict__ Aj)
{
    __shared__ float comb[HH];
    const int row = blockIdx.x;        // 0 .. B*N-1
    const int t   = threadIdx.x;       // 0 .. 63
    const float* c4 = coords + row * 4;
    float acc = feat[row * HH + t] + bp[t];
    acc += c4[0] * Wp[0 * HH + t] + c4[1] * Wp[1 * HH + t]
         + c4[2] * Wp[2 * HH + t] + c4[3] * Wp[3 * HH + t];
    comb[t] = acc;
    __syncthreads();
    float ai = b1[t];
    float aj = 0.f;
#pragma unroll
    for (int h = 0; h < HH; ++h) {
        const float cv = comb[h];
        ai += cv * W1[h * HH + t];
        aj += cv * W1[(HH + h) * HH + t];
    }
    Ai[row * HH + t] = ai;
    Aj[row * HH + t] = aj;
}

// ---------------------------------------------------------------------------
// Kernel 0b: pack W2 (f32 [64][64], h @ W2 so rows = K) into the exact
// v_wmma_f32_16x16x32_f16 B-matrix image used here (contiguous k, as A), per lane, so the main
// kernel loads each 16x16 B tile's 8 VGPRs with one 32-byte vector load.
//   layout: pack[((kt*4+nt)*32 + lane)*8 + v] = {f16 W2[k][n], f16 W2[k+1][n]}
//   with k = kt*32 + (lane>>4)*16 + 2v,  n = nt*16 + (lane&15).
// ---------------------------------------------------------------------------
__global__ __launch_bounds__(256) void sg_packW2(
    const float* __restrict__ W2, unsigned int* __restrict__ pack)
{
    const int t = blockIdx.x * 256 + threadIdx.x;   // 0 .. 2047
    if (t >= 2 * 4 * 32 * 8) return;
    const int v    = t & 7;
    const int lane = (t >> 3) & 31;
    const int nt   = (t >> 8) & 3;
    const int kt   = t >> 10;
    const int l16  = lane & 15;
    const int hi   = lane >> 4;
    const int k    = kt * 32 + hi * 16 + 2 * v;
    const int n    = nt * 16 + l16;
    const _Float16 lo = (_Float16)W2[k * HH + n];
    const _Float16 hh = (_Float16)W2[(k + 1) * HH + n];
    unsigned int w;
    unsigned short lob = __builtin_bit_cast(unsigned short, lo);
    unsigned short hib = __builtin_bit_cast(unsigned short, hh);
    w = (unsigned int)lob | ((unsigned int)hib << 16);
    pack[t] = w;
}

// ---------------------------------------------------------------------------
// Kernel 1: per wave32: 16 (i,j)-pair rows x H=64 through
//   h2 = relu( relu(Ai'[i]+Aj[j]) @ W2 + b2 ) ; logits = h2 @ W3 + b3 ;
//   softmax / argmax / diag mask.
//   GEMM1 via v_wmma_f32_16x16x32_f16 (4 N-tiles x 2 K-steps).
// ---------------------------------------------------------------------------
__global__ __launch_bounds__(128) void sg_pair_kernel(
    const float* __restrict__ Ai, const float* __restrict__ Aj,
    const v16h* __restrict__ W2pack, const float* __restrict__ b2,
    const float* __restrict__ W3, const float* __restrict__ b3,
    float* __restrict__ probs, float* __restrict__ relf, float* __restrict__ conf)
{
    const int lane = threadIdx.x & 31;
    const int l16  = lane & 15;
    const int hi   = lane >> 4;
    const int wave = blockIdx.x * 4 + (threadIdx.x >> 5);  // 0 .. 32767
    const int b    = wave >> 10;          // / (8*128)
    const int rem  = wave & 1023;
    const int i    = rem >> 3;
    const int j0   = (rem & 7) << 4;

    // ---- B matrices: one 32-byte vector load per 16x16 tile (pre-packed).
    v16h Bm[2][4];
#pragma unroll
    for (int kt = 0; kt < 2; ++kt)
#pragma unroll
        for (int nt = 0; nt < 4; ++nt)
            Bm[kt][nt] = W2pack[(kt * 4 + nt) * 32 + lane];

    // ---- Build h0 = relu(Ai'[i] + Aj[j0+l16]) directly in A-matrix layout.
    // A layout used here (contiguous k, same as W2pack): lane(hi,l16) row m=l16; element e -> K=hi*16+e.
    const float* aiRow = Ai + (b * NN + i) * HH;
    const float* ajRow = Aj + (b * NN + (j0 + l16)) * HH;
    v16h a0, a1;
#pragma unroll
    for (int ch = 0; ch < 4; ++ch) {
        const int cb = (ch >> 1) * 32 + hi * 16 + (ch & 1) * 8;  // 32B-aligned chunk base
        const float4 x0 = *(const float4*)(aiRow + cb);
        const float4 x1 = *(const float4*)(aiRow + cb + 4);
        const float4 y0 = *(const float4*)(ajRow + cb);
        const float4 y1 = *(const float4*)(ajRow + cb + 4);
        v16h* dst = (ch < 2) ? &a0 : &a1;
        const int o = (ch & 1) * 8;
        (*dst)[o + 0] = reluh(x0.x + y0.x);
        (*dst)[o + 1] = reluh(x0.y + y0.y);
        (*dst)[o + 2] = reluh(x0.z + y0.z);
        (*dst)[o + 3] = reluh(x0.w + y0.w);
        (*dst)[o + 4] = reluh(x1.x + y1.x);
        (*dst)[o + 5] = reluh(x1.y + y1.y);
        (*dst)[o + 6] = reluh(x1.z + y1.z);
        (*dst)[o + 7] = reluh(x1.w + y1.w);
    }

    // ---- GEMM1: [16x64] = h0[16x64] @ W2[64x64], two K=32 WMMA per N-tile.
    v8f acc[4];
#pragma unroll
    for (int nt = 0; nt < 4; ++nt) {
        v8f c = {};
        c = __builtin_amdgcn_wmma_f32_16x16x32_f16(false, a0, false, Bm[0][nt],
                                                   (short)0, c, false, false);
        c = __builtin_amdgcn_wmma_f32_16x16x32_f16(false, a1, false, Bm[1][nt],
                                                   (short)0, c, false, false);
        acc[nt] = c;
    }

    // ---- bias + relu. C layout: vgpr v -> row m=hi*8+v, lane l16 -> col n=nt*16+l16.
#pragma unroll
    for (int nt = 0; nt < 4; ++nt) {
        const float bias = b2[nt * 16 + l16];
#pragma unroll
        for (int v = 0; v < 8; ++v) {
            const float h2 = acc[nt][v] + bias;
            acc[nt][v] = h2 > 0.f ? h2 : 0.f;
        }
    }

    // ---- W3 columns owned by this lane (n = nt*16 + l16).
    float w3l[4][3];
#pragma unroll
    for (int nt = 0; nt < 4; ++nt) {
        const int n = nt * 16 + l16;
#pragma unroll
        for (int c = 0; c < 3; ++c) w3l[nt][c] = W3[n * 3 + c];
    }
    const float b30 = b3[0], b31 = b3[1], b32 = b3[2];

    const int outRowBase = (b * NN + i) * NN;

    // ---- logits = h2 @ W3: per-lane partials + 16-lane butterfly reduce
    //      (each half-wave owns disjoint rows: hi*8 + v).
#pragma unroll
    for (int v = 0; v < 8; ++v) {
        float p0 = 0.f, p1 = 0.f, p2 = 0.f;
#pragma unroll
        for (int nt = 0; nt < 4; ++nt) {
            const float hv = acc[nt][v];
            p0 += hv * w3l[nt][0];
            p1 += hv * w3l[nt][1];
            p2 += hv * w3l[nt][2];
        }
#pragma unroll
        for (int m = 8; m >= 1; m >>= 1) {   // stays inside the 16-lane half
            p0 += __shfl_xor(p0, m, 32);
            p1 += __shfl_xor(p1, m, 32);
            p2 += __shfl_xor(p2, m, 32);
        }
        if (l16 == v) {                       // one writer lane per half per row
            const float lg0 = p0 + b30, lg1 = p1 + b31, lg2 = p2 + b32;
            const float mx = fmaxf(lg0, fmaxf(lg1, lg2));
            const float e0 = __expf(lg0 - mx);
            const float e1 = __expf(lg1 - mx);
            const float e2 = __expf(lg2 - mx);
            const float inv = 1.f / (e0 + e1 + e2);
            const float q0 = e0 * inv, q1 = e1 * inv, q2 = e2 * inv;
            const int j  = j0 + hi * 8 + v;
            const int ri = outRowBase + j;
            const int pi = ri * 3;
            probs[pi]     = q0;
            probs[pi + 1] = q1;
            probs[pi + 2] = q2;
            int   idx  = 0;
            float best = q0;
            if (q1 > best) { best = q1; idx = 1; }
            if (q2 > best) { best = q2; idx = 2; }
            if (i == j) { relf[ri] = -1.0f; conf[ri] = 0.0f; }
            else        { relf[ri] = (float)idx; conf[ri] = best; }
        }
    }
}

extern "C" void kernel_launch(void* const* d_in, const int* in_sizes, int n_in,
                              void* d_out, int out_size, void* d_ws, size_t ws_size,
                              hipStream_t stream) {
    (void)in_sizes; (void)n_in; (void)out_size; (void)ws_size;
    const float* feat   = (const float*)d_in[0];
    const float* coords = (const float*)d_in[1];
    /* d_in[2] = obj_classes: unused by the reference computation */
    const float* Wp = (const float*)d_in[3];
    const float* bp = (const float*)d_in[4];
    const float* W1 = (const float*)d_in[5];
    const float* b1 = (const float*)d_in[6];
    const float* W2 = (const float*)d_in[7];
    const float* b2 = (const float*)d_in[8];
    const float* W3 = (const float*)d_in[9];
    const float* b3 = (const float*)d_in[10];

    float* Ai = (float*)d_ws;                 // [B*N, H]  (1 MB)
    float* Aj = Ai + BB * NN * HH;            // [B*N, H]  (1 MB)
    unsigned int* W2pack = (unsigned int*)(Aj + BB * NN * HH);  // 8 KB, 32B aligned

    float* out   = (float*)d_out;
    float* probs = out;                       // [B,N,N,3]
    float* relf  = out + BB * NN * NN * 3;    // [B,N,N]
    float* conf  = relf + BB * NN * NN;       // [B,N,N]

    sg_prelayer<<<BB * NN, 64, 0, stream>>>(feat, coords, Wp, bp, W1, b1, Ai, Aj);
    sg_packW2<<<8, 256, 0, stream>>>(W2, W2pack);

    // 32768 tiles (b * i * 8 j-blocks), 1 wave each, 4 waves per block.
    sg_pair_kernel<<<(BB * NN * (NN / 16)) / 4, 128, 0, stream>>>(
        Ai, Aj, (const v16h*)W2pack, b2, W3, b3, probs, relf, conf);
}
